// MHSA3D_45689862095462
// MI455X (gfx1250) — matrix operand order repaired and checked by reading, not compiled, not run on hardware
//
#include <hip/hip_runtime.h>

// MI455X / gfx1250: wave32, WMMA 16x16x32 bf16 (f32 accum), async-LDS + TDM.
typedef __attribute__((ext_vector_type(16))) __bf16 v16bf;
typedef __attribute__((ext_vector_type(8)))  float  v8f;
typedef unsigned v4u __attribute__((ext_vector_type(4)));
typedef unsigned v8u __attribute__((ext_vector_type(8)));

// B=16, C=256, F=4, H=32, W=32, heads=4, dh=64, HW=1024; 256 slabs of [64x1024].

// ---- CDNA5 async global->LDS copy (ASYNCcnt), 16 bytes per lane ------------
__device__ __forceinline__ void async_copy16(void* lds, const void* gmem) {
  const unsigned lds_off = (unsigned)(size_t)lds;  // flat->LDS offset (addr[31:0])
  asm volatile("global_load_async_to_lds_b128 %0, %1, off"
               :
               : "v"(lds_off), "v"((unsigned long long)(size_t)gmem)
               : "memory");
}
__device__ __forceinline__ void async_wait0() {
  asm volatile("s_wait_asynccnt 0" ::: "memory");
}

// ---- CDNA5 Tensor Data Mover: 2D bf16 tile, global -> LDS (TENSORcnt) ------
// D# per ISA ch.8: group0 = {count=1, lds_addr, global_addr[56:0], type=2};
// group1 = {data_size=2B, tensor_dim0/1, tile_dim0/1, tensor_dim0_stride}.
// 2-operand form => VADDR2/VADDR3 = NULL (tensor <= 2D).
__device__ __forceinline__ void tdm_load_2d_bf16(
    void* lds, const void* gmem, unsigned tile_w, unsigned tile_h,
    unsigned row_stride, unsigned tensor_w, unsigned tensor_h) {
  const unsigned long long ga = (unsigned long long)(size_t)gmem;
  v4u g0;
  g0[0] = 1u;                                   // count=1 (user descriptor)
  g0[1] = (unsigned)(size_t)lds;                // lds_addr (bytes)
  g0[2] = (unsigned)ga;                         // global_addr[31:0]
  g0[3] = (unsigned)((ga >> 32) & 0x01FFFFFFu)  // global_addr[56:32]
          | (2u << 30);                         // type=2 ("image")
  v8u g1;
  g1[0] = 1u << 16;                             // data_size=1 (2 bytes/elem)
  g1[1] = (tensor_w & 0xFFFFu) << 16;           // tensor_dim0[15:0]
  g1[2] = (tensor_w >> 16) | ((tensor_h & 0xFFFFu) << 16);  // d0 hi | d1 lo
  g1[3] = (tensor_h >> 16) | (tile_w << 16);    // d1 hi | tile_dim0
  g1[4] = tile_h;                               // tile_dim1 (tile_dim2=0 -> 2D)
  g1[5] = row_stride;                           // tensor_dim0_stride[31:0]
  g1[6] = 0;                                    // stride hi | dim1_stride lo
  g1[7] = 0;
  asm volatile("tensor_load_to_lds %0, %1" :: "s"(g0), "s"(g1) : "memory");
}

// ---- Fragment builders per CDNA5 ISA 7.12.2 VGPR layouts (wave32) ----------
template <typename SRC>
__device__ __forceinline__ v16bf make_a_frag(SRC src) {  // A 16x32 bf16
  const int lane = threadIdx.x & 31;
  const int m  = lane & 15;
  const int kh = (lane >> 4) << 3;
  v16bf a;
#pragma unroll
  for (int j = 0; j < 4; ++j) {
    const int k0 = kh + 2 * j;
    a[2 * j + 0] = src(m, k0 + 0);
    a[2 * j + 1] = src(m, k0 + 1);
    a[2 * j + 8] = src(m, 16 + k0 + 0);
    a[2 * j + 9] = src(m, 16 + k0 + 1);
  }
  return a;
}
template <typename SRC>
__device__ __forceinline__ v16bf make_b_frag(SRC src) {  // B 32x16 bf16
  const int lane = threadIdx.x & 31;
  const int n  = lane & 15;
  const int kb = (lane >> 4) << 3;
  v16bf b;
#pragma unroll
  for (int v = 0; v < 8; ++v) {
    b[2 * v + 0] = src(kb + 2 * v + (v >> 2) * 8 + 0, n);
    b[2 * v + 1] = src(kb + 2 * v + (v >> 2) * 8 + 1, n);
  }
  return b;
}
// C/D 16x16 f32: VGPR v holds (m = v + 8*(lane>>4), n = lane&15).

// ---------------------------------------------------------------------------
// Kernel 1: QKV projection, double-buffered async-LDS pipeline.
// Block = 256 thr (8 waves), 32x64 output tile; panels for k-step i+1 are in
// flight (ASYNCcnt) while WMMAs consume k-step i.
// grid = (Nblk=64, Mblk=8, B*3=48).
// ---------------------------------------------------------------------------
__global__ void __launch_bounds__(256)
mhsa3d_qkv_proj(const float* __restrict__ x,
                const float* __restrict__ wq, const float* __restrict__ wk,
                const float* __restrict__ wv,
                __bf16* __restrict__ yq, __bf16* __restrict__ yk,
                __bf16* __restrict__ yv) {
  __shared__ float Ws[2][32][32];  // 2 x 4 KB, rows 128B-aligned (async b128)
  __shared__ float Xs[2][32][64];  // 2 x 8 KB, rows 256B-aligned

  const int tid  = threadIdx.x;
  const int wave = tid >> 5;
  const int lane = tid & 31;
  const int n0 = blockIdx.x * 64;
  const int m0 = blockIdx.y * 32;
  const int b   = blockIdx.z / 3;
  const int mat = blockIdx.z % 3;
  const float* w = (mat == 0) ? wq : (mat == 1) ? wk : wv;
  __bf16*      y = (mat == 0) ? yq : (mat == 1) ? yk : yv;
  const size_t xbase = (size_t)b * 256 * 4096;
  const int mrel = (wave >> 2) * 16;
  const int nrel = (wave & 3) * 16;

  const int r  = tid >> 3, c  = (tid & 7) * 4;   // W panel: 256 xfers
  const int r2 = tid >> 4, c2 = (tid & 15) * 4;  // X panel: 512 xfers

  auto stage = [&](int buf, int k0) {
    async_copy16(&Ws[buf][r][c], &w[(size_t)(m0 + r) * 256 + k0 + c]);
    async_copy16(&Xs[buf][r2][c2],
                 &x[xbase + (size_t)(k0 + r2) * 4096 + n0 + c2]);
    async_copy16(&Xs[buf][r2 + 16][c2],
                 &x[xbase + (size_t)(k0 + r2 + 16) * 4096 + n0 + c2]);
  };

  stage(0, 0);
  async_wait0();
  __syncthreads();

  v8f acc = {};
  int cur = 0;
  for (int k0 = 0; k0 < 256; k0 += 32, cur ^= 1) {
    if (k0 + 32 < 256) stage(cur ^ 1, k0 + 32);  // overlap DMA with compute
    v16bf a = make_a_frag([&](int m, int k) -> __bf16 {
      return (__bf16)Ws[cur][mrel + m][k];
    });
    v16bf bx = make_b_frag([&](int k, int n) -> __bf16 {
      return (__bf16)Xs[cur][k][nrel + n];
    });
    acc = __builtin_amdgcn_wmma_f32_16x16x32_bf16(false, a, false, bx,
                                                  (short)0, acc, false, false);
    async_wait0();
    __syncthreads();
  }
  const int cm = 8 * (lane >> 4), cn = lane & 15;
#pragma unroll
  for (int v = 0; v < 8; ++v)
    y[xbase + (size_t)(m0 + mrel + cm + v) * 4096 + (n0 + nrel + cn)] =
        (__bf16)acc[v];
}

// ---------------------------------------------------------------------------
// Kernel 2: attention for one slab (b,h,f') and one 64-row query block.
// K/Q/V panels are DMA'd to LDS by the Tensor Data Mover (one wave issues,
// TENSORcnt wait + barrier publishes). LDS: E[64][1024] f32 + Qt/P[64][72]
// bf16 + K/Q panels [64][128] bf16 (V aliases K) + scratch ~= 307 KB.
// grid = (16 qblocks, 256 slabs), block = 256 (8 waves).
// ---------------------------------------------------------------------------
__global__ void __launch_bounds__(256)
mhsa3d_attention(const __bf16* __restrict__ Qn, const __bf16* __restrict__ Kn,
                 const __bf16* __restrict__ Vn,
                 const float* __restrict__ rel_h, const float* __restrict__ rel_w,
                 const float* __restrict__ rel_t, float* __restrict__ out) {
  extern __shared__ char smem[];
  float*  E      = (float*)smem;                       // [64][1024] f32
  __bf16* Qt     = (__bf16*)(smem + 262144);           // [64][72]   (Q^T)
  __bf16* Pblk   = (__bf16*)(smem + 262144 + 9216);    // [64][72]
  __bf16* Kp     = (__bf16*)(smem + 280576);           // [64][128]  (TDM dst)
  __bf16* Qp     = (__bf16*)(smem + 296960);           // [64][128]  (TDM dst)
  __bf16* Vp     = Kp;                                 // phase-3 alias
  float*  red    = (float*)(smem + 313344);            // [256]
  float*  rowinv = red + 256;                          // [64]

  const int tid  = threadIdx.x;
  const int wave = tid >> 5;
  const int lane = tid & 31;
  const int slab = blockIdx.y;  // b*16 + h*4 + f'
  const int h    = (slab >> 2) & 3;
  const int fp   = slab & 3;
  const int qbase = blockIdx.x * 64;
  const size_t sb = (size_t)slab * 65536;

  // Phase 0: Qt[i][d] (coalesced 8B reads, transposed ds stores) and P[i][d].
  for (int idx = tid; idx < 64 * 16; idx += 256) {
    const int d  = idx >> 4;
    const int i4 = (idx & 15) * 4;
    uint2 q4 = *(const uint2*)&Qn[sb + (size_t)d * 1024 + qbase + i4];
    const __bf16* q = (const __bf16*)&q4;
#pragma unroll
    for (int j = 0; j < 4; ++j) Qt[(i4 + j) * 72 + d] = q[j];
  }
  for (int idx = tid; idx < 64 * 64; idx += 256) {
    const int i = idx >> 6, d = idx & 63;
    const int p = qbase + i;
    const int hh = p >> 5, ww = p & 31;
    const int hd = h * 64 + d;
    Pblk[i * 72 + d] = (__bf16)(rel_h[hd * 32 + ww] + rel_w[hd * 32 + hh] +
                                rel_t[hd * 4 + fp]);
  }
  __syncthreads();

  // Phase 1: E[i][j] = 0.125*( sum_d Qt[i,d]K[d,j] + sum_d P[i,d]Q[d,j] ).
  for (int jc = 0; jc < 1024; jc += 128) {
    if (wave == 0) {  // TDM: two 64x128 bf16 tiles, row stride 1024 elems
      tdm_load_2d_bf16(Kp, &Kn[sb + jc], 128, 64, 1024, 1024, 64);
      tdm_load_2d_bf16(Qp, &Qn[sb + jc], 128, 64, 1024, 1024, 64);
      __builtin_amdgcn_s_wait_tensorcnt(0);
    }
    __syncthreads();
    for (int t = wave; t < 32; t += 8) {   // 4 mtiles x 8 jtiles
      const int i0 = (t >> 3) * 16;
      const int j0 = (t & 7) * 16;
      v8f acc = {};
#pragma unroll
      for (int ks = 0; ks < 64; ks += 32) {  // content-content
        v16bf a = make_a_frag([&](int m, int k) -> __bf16 {
          return Qt[(i0 + m) * 72 + ks + k];
        });
        v16bf bk = make_b_frag([&](int k, int n) -> __bf16 {
          return Kp[(ks + k) * 128 + j0 + n];
        });
        acc = __builtin_amdgcn_wmma_f32_16x16x32_bf16(false, a, false, bk,
                                                      (short)0, acc, false, false);
      }
#pragma unroll
      for (int ks = 0; ks < 64; ks += 32) {  // content-position
        v16bf a = make_a_frag([&](int m, int k) -> __bf16 {
          return Pblk[(i0 + m) * 72 + ks + k];
        });
        v16bf bq = make_b_frag([&](int k, int n) -> __bf16 {
          return Qp[(ks + k) * 128 + j0 + n];
        });
        acc = __builtin_amdgcn_wmma_f32_16x16x32_bf16(false, a, false, bq,
                                                      (short)0, acc, false, false);
      }
      const int cm = 8 * (lane >> 4), cn = lane & 15;
#pragma unroll
      for (int v = 0; v < 8; ++v)
        E[(i0 + cm + v) * 1024 + jc + j0 + cn] = acc[v] * 0.125f;
    }
    __syncthreads();
  }

  // Phase 2: row softmax over 1024 columns (4 threads per row).
  {
    const int g = tid >> 2, sub = tid & 3;
    float* row = E + g * 1024 + sub * 256;
    float mx = -3.402823e38f;
    for (int j = 0; j < 256; ++j) mx = fmaxf(mx, row[j]);
    red[tid] = mx;
    __syncthreads();
    const float m4 = fmaxf(fmaxf(red[g * 4 + 0], red[g * 4 + 1]),
                           fmaxf(red[g * 4 + 2], red[g * 4 + 3]));
    float s = 0.f;
    for (int j = 0; j < 256; ++j) {
      float pj = __expf(row[j] - m4);
      row[j] = pj;
      s += pj;
    }
    __syncthreads();
    red[tid] = s;
    __syncthreads();
    if (sub == 0)
      rowinv[g] = 1.0f / (red[g * 4 + 0] + red[g * 4 + 1] +
                          red[g * 4 + 2] + red[g * 4 + 3]);
    __syncthreads();
  }

  // Phase 3: O[i][d] = sum_j attn[i][j]*V[d][j]; V tile via TDM per chunk.
  const int i0a = (wave >> 2) * 16;  // tile t = wave
  const int i0b = i0a + 32;          // tile t = wave + 8
  const int d0  = (wave & 3) * 16;
  v8f acc0 = {}, acc1 = {};
  for (int jc = 0; jc < 1024; jc += 128) {
    if (wave == 0) {
      tdm_load_2d_bf16(Vp, &Vn[sb + jc], 128, 64, 1024, 1024, 64);
      __builtin_amdgcn_s_wait_tensorcnt(0);
    }
    __syncthreads();
#pragma unroll
    for (int jj = 0; jj < 128; jj += 32) {
      v16bf bv = make_b_frag([&](int k, int n) -> __bf16 {
        return Vp[(d0 + n) * 128 + jj + k];  // V^T fragment
      });
      v16bf a0 = make_a_frag([&](int m, int k) -> __bf16 {
        return (__bf16)(E[(i0a + m) * 1024 + jc + jj + k] * rowinv[i0a + m]);
      });
      acc0 = __builtin_amdgcn_wmma_f32_16x16x32_bf16(false, a0, false, bv,
                                                     (short)0, acc0, false, false);
      v16bf a1 = make_a_frag([&](int m, int k) -> __bf16 {
        return (__bf16)(E[(i0b + m) * 1024 + jc + jj + k] * rowinv[i0b + m]);
      });
      acc1 = __builtin_amdgcn_wmma_f32_16x16x32_bf16(false, a1, false, bv,
                                                     (short)0, acc1, false, false);
    }
    __syncthreads();
  }
  const int cm = 8 * (lane >> 4), cn = lane & 15;
#pragma unroll
  for (int v = 0; v < 8; ++v) {
    out[sb + (size_t)(d0 + cn) * 1024 + qbase + i0a + cm + v] = acc0[v];
    out[sb + (size_t)(d0 + cn) * 1024 + qbase + i0b + cm + v] = acc1[v];
  }
}

// ---------------------------------------------------------------------------
extern "C" void kernel_launch(void* const* d_in, const int* in_sizes, int n_in,
                              void* d_out, int out_size, void* d_ws,
                              size_t ws_size, hipStream_t stream) {
  (void)in_sizes; (void)n_in; (void)out_size; (void)ws_size;
  const float* x    = (const float*)d_in[0];
  const float* wq   = (const float*)d_in[1];
  const float* wk   = (const float*)d_in[2];
  const float* wv   = (const float*)d_in[3];
  const float* relh = (const float*)d_in[4];
  const float* relw = (const float*)d_in[5];
  const float* relt = (const float*)d_in[6];
  float* out = (float*)d_out;

  const size_t elems = (size_t)256 * 65536;  // per bf16 Q/K/V matrix
  __bf16* Qn = (__bf16*)d_ws;
  __bf16* Kn = Qn + elems;
  __bf16* Vn = Kn + elems;

  dim3 pg(64, 8, 48), pb(256);
  hipLaunchKernelGGL(mhsa3d_qkv_proj, pg, pb, 0, stream,
                     x, wq, wk, wv, Qn, Kn, Vn);

  const size_t shmem = 262144 + 9216 + 9216 + 16384 + 16384 + 1024 + 256;
  dim3 ag(16, 256), ab(256);
  hipLaunchKernelGGL(mhsa3d_attention, ag, ab, shmem, stream,
                     Qn, Kn, Vn, relh, relw, relt, out);
}
